// MaskGITTokenizer_17085379903722
// MI455X (gfx1250) — matrix operand order repaired and checked by reading, not compiled, not run on hardware
//
#include <hip/hip_runtime.h>
#include <stdint.h>

// ---------------------------------------------------------------------------
// CDNA5 (gfx1250) MaskGIT tokenizer:
//  - implicit-GEMM convs on v_wmma_f32_16x16x32_f16 (f32 accumulate)
//  - double-buffered TDM (tensor_load_to_lds) weight pipeline, TENSORcnt-overlapped
//  - async global->LDS staging (ASYNCcnt) in the VQ distance kernel
//  - f16 activations everywhere (HBM-bound model: halves traffic vs f32)
// ---------------------------------------------------------------------------

typedef __attribute__((ext_vector_type(16))) _Float16 v16h;
typedef __attribute__((ext_vector_type(8)))  _Float16 v8h;
typedef __attribute__((ext_vector_type(8)))  float    v8f;
typedef __attribute__((ext_vector_type(4)))  unsigned int u32x4;
typedef __attribute__((ext_vector_type(8)))  int      i32x8;
typedef __attribute__((ext_vector_type(4)))  int      i32x4;

// ---------------- elementwise helpers ----------------

__global__ void k_f32_to_f16(const float* __restrict__ s, _Float16* __restrict__ d, int n) {
  int i = blockIdx.x * 256 + threadIdx.x;
  if (i < n) d[i] = (_Float16)s[i];
}

// GroupNorm statistics: one block per (batch, group); 32 groups fixed.
__global__ void k_gn_stats(const _Float16* __restrict__ x, float* __restrict__ stats,
                           int C, int HW) {
  int b = blockIdx.x >> 5, g = blockIdx.x & 31;
  int cg = C >> 5;
  size_t n = (size_t)cg * HW;
  const _Float16* base = x + ((size_t)b * C + (size_t)g * cg) * HW;
  float s = 0.f, s2 = 0.f;
  for (size_t i = threadIdx.x; i < n; i += 256) {
    float v = (float)base[i];
    s += v; s2 += v * v;
  }
  __shared__ float sh[512];
  sh[threadIdx.x] = s; sh[256 + threadIdx.x] = s2;
  __syncthreads();
  for (int st = 128; st > 0; st >>= 1) {
    if (threadIdx.x < st) {
      sh[threadIdx.x] += sh[threadIdx.x + st];
      sh[256 + threadIdx.x] += sh[256 + threadIdx.x + st];
    }
    __syncthreads();
  }
  if (threadIdx.x == 0) {
    float inv = 1.f / (float)n;
    float mu = sh[0] * inv;
    float var = sh[256] * inv - mu * mu;
    stats[blockIdx.x * 2] = mu;
    stats[blockIdx.x * 2 + 1] = rsqrtf(var + 1e-6f);
  }
}

// Fused GroupNorm-apply + SiLU, f16 in/out.
__global__ void k_gn_silu(const _Float16* __restrict__ x, _Float16* __restrict__ y,
                          const float* __restrict__ gam, const float* __restrict__ bet,
                          const float* __restrict__ stats, int C, int HW, size_t total) {
  int cg = C >> 5;
  for (size_t i = blockIdx.x * (size_t)256 + threadIdx.x; i < total;
       i += (size_t)gridDim.x * 256) {
    int c = (int)((i / (size_t)HW) % C);
    int b = (int)(i / ((size_t)HW * C));
    int gi = (b << 5) | (c / cg);
    float mu = stats[gi * 2], rs = stats[gi * 2 + 1];
    float v = ((float)x[i] - mu) * rs * gam[c] + bet[c];
    y[i] = (_Float16)(v / (1.f + __expf(-v)));
  }
}

__global__ void k_avgpool(const _Float16* __restrict__ x, _Float16* __restrict__ y,
                          int H, int W, size_t total) {
  int Wo = W >> 1, Ho = H >> 1;
  for (size_t i = blockIdx.x * (size_t)256 + threadIdx.x; i < total;
       i += (size_t)gridDim.x * 256) {
    int xo = (int)(i % Wo);
    size_t t = i / Wo;
    int yo = (int)(t % Ho);
    size_t bc = t / Ho;
    const _Float16* p = x + (bc * H + (size_t)yo * 2) * W + (size_t)xo * 2;
    y[i] = (_Float16)(0.25f * ((float)p[0] + (float)p[1] + (float)p[W] + (float)p[W + 1]));
  }
}

__global__ void k_up2(const _Float16* __restrict__ x, _Float16* __restrict__ y,
                      int H, int W, size_t total) {
  int Wo = W << 1, Ho = H << 1;
  for (size_t i = blockIdx.x * (size_t)256 + threadIdx.x; i < total;
       i += (size_t)gridDim.x * 256) {
    int xo = (int)(i % Wo);
    size_t t = i / Wo;
    int yo = (int)(t % Ho);
    size_t bc = t / Ho;
    y[i] = x[(bc * H + (yo >> 1)) * W + (xo >> 1)];
  }
}

// ---------------- implicit-GEMM conv with WMMA + pipelined TDM ----------------
// Block tile: 64 couts x 32 pixels. 8 waves = 4(M) x 2(N) subtiles of 16x16.
// K consumed 64 at a time (2 x v_wmma_f32_16x16x32_f16 per wave per stage).
// Weight tiles [64 x 64] f16 are DMA'd by the Tensor Data Mover into a
// double-buffered LDS region: A(c+1) is issued BEFORE waiting on A(c)
// (TENSORcnt is in-order per wave, so s_wait_tensorcnt 1 completes A(c)),
// overlapping the DMA with the im2col gather and the previous compute.
__launch_bounds__(256)
__global__ void k_conv_wmma(const _Float16* __restrict__ in, const _Float16* __restrict__ wgt,
                            const float* __restrict__ bias, const _Float16* __restrict__ resid,
                            _Float16* __restrict__ out16, float* __restrict__ out32,
                            int Cin, int H, int W, int Cout, int Ksz) {
  const int Ks2 = Ksz * Ksz;
  const int pad = (Ksz - 1) >> 1;
  const int Ktot = Cin * Ks2;
  const int nchunks = (Ktot + 63) >> 6;
  const int nXC = W >> 5;
  const int y   = blockIdx.x / nXC;
  const int x0  = (blockIdx.x % nXC) << 5;
  const int co0 = blockIdx.y << 6;
  const int b   = blockIdx.z;

  __shared__ __align__(16) _Float16 lds_a[2][64][64];   // weights [cout][k]
  __shared__ __align__(16) _Float16 lds_b[2][32][64];   // im2col  [pix][k]

  const int tid = threadIdx.x, wave = tid >> 5, lane = tid & 31;
  const int msub = wave >> 1, nsub = wave & 1;

  // TDM descriptor for weight tile [64 rows x 64 K] at chunk c into buffer buf.
  auto issue_tdm = [&](int c, int buf) {
    int k0 = c << 6;
    unsigned lds_off = (unsigned)(uintptr_t)(&lds_a[buf][0][0]);
    unsigned long long ga =
        (unsigned long long)(uintptr_t)(wgt + (size_t)co0 * Ktot + k0);
    unsigned td0 = (unsigned)(Ktot - k0);                      // remaining K
    unsigned td1 = (unsigned)(Cout > co0 ? Cout - co0 : 0);    // remaining couts
    u32x4 g0;
    g0[0] = 1u;                                   // count=1, user descriptor
    g0[1] = lds_off;                              // lds_addr
    g0[2] = (unsigned)ga;                         // global_addr lo
    g0[3] = (unsigned)((ga >> 32) & 0x01FFFFFFu) | 0x80000000u;  // hi + type=2
    i32x8 g1;
    g1[0] = (int)0x00010000u;                     // data_size=1 (2 bytes)
    g1[1] = (int)((td0 & 0xFFFFu) << 16);         // tensor_dim0[15:0]
    g1[2] = (int)((td0 >> 16) | ((td1 & 0xFFFFu) << 16));
    g1[3] = (int)((td1 >> 16) | (64u << 16));     // td1 hi | tile_dim0=64
    g1[4] = 64;                                   // tile_dim1=64, tile_dim2=0
    g1[5] = (int)(unsigned)Ktot;                  // tensor_dim0_stride lo32
    g1[6] = 0;
    g1[7] = 0;
    i32x4 zz; zz[0] = 0; zz[1] = 0; zz[2] = 0; zz[3] = 0;
    i32x8 z8;
#pragma unroll
    for (int q = 0; q < 8; ++q) z8[q] = 0;
    __builtin_amdgcn_tensor_load_to_lds(g0, g1, zz, zz, z8, 0);
  };

  if (tid < 32) issue_tdm(0, 0);

  v8f acc = {0.f, 0.f, 0.f, 0.f, 0.f, 0.f, 0.f, 0.f};

  for (int c = 0; c < nchunks; ++c) {
    const int buf = c & 1;
    const int k0 = c << 6;

    // --- B tile: im2col gather (overlaps in-flight TDM) ---
    {
      int p = tid & 31;
      int kk = (tid >> 5) << 3;      // 8 k's per thread
      int xx0 = x0 + p;
      if (c + 1 < nchunks) {
        int kn = k0 + 64;
        int cinn = kn / Ks2;
        if (cinn < Cin)
          __builtin_prefetch(&in[(((size_t)b * Cin + cinn) * H + y) * W + x0], 0, 0);
      }
      v8h pv;
#pragma unroll
      for (int j = 0; j < 8; ++j) {
        int k = k0 + kk + j;
        _Float16 v = (_Float16)0.f;
        if (k < Ktot) {
          int cin = k / Ks2;
          int rem = k - cin * Ks2;
          int dy = rem / Ksz - pad;
          int dx = rem - (rem / Ksz) * Ksz - pad;
          int yy = y + dy, xx = xx0 + dx;
          if ((unsigned)yy < (unsigned)H && (unsigned)xx < (unsigned)W)
            v = in[(((size_t)b * Cin + cin) * H + yy) * W + xx];
        }
        pv[j] = v;
      }
      *(v8h*)&lds_b[buf][p][kk] = pv;
    }

    // --- pipeline: issue next A tile, then wait for current one ---
    if (tid < 32) {
      if (c + 1 < nchunks) {
        issue_tdm(c + 1, buf ^ 1);
        __builtin_amdgcn_s_wait_tensorcnt((short)1);  // A(c) done, A(c+1) in flight
      } else {
        __builtin_amdgcn_s_wait_tensorcnt((short)0);  // last chunk: drain
      }
    }
    __syncthreads();

    // --- 2 x WMMA over this K=64 stage (contiguous k = 16*(lane/16) + s in both A and B) ---
    const int arow = (msub << 4) + (lane & 15);
    const int akb = (lane < 16) ? 0 : 16;
    const int bcol = (nsub << 4) + (lane & 15);
    const int bkb = (lane < 16) ? 0 : 16;
#pragma unroll
    for (int h = 0; h < 2; ++h) {
      const int kb = h << 5;
      union { v16h v; v8h h2[2]; } au;
      au.h2[0] = *(const v8h*)&lds_a[buf][arow][kb + akb];
      au.h2[1] = *(const v8h*)&lds_a[buf][arow][kb + akb + 8];
      v16h bv = *(const v16h*)&lds_b[buf][bcol][kb + bkb];
      acc = __builtin_amdgcn_wmma_f32_16x16x32_f16(false, au.v, false, bv,
                                                   (short)0, acc, false, false);
    }
    __syncthreads();
  }

  // --- epilogue: bias + fused residual add, f16 and/or f32 stores ---
  int col = lane & 15;
  int xo = x0 + (nsub << 4) + col;
  int rbase = co0 + (msub << 4) + ((lane >= 16) ? 8 : 0);
#pragma unroll
  for (int r = 0; r < 8; ++r) {
    int co = rbase + r;
    if (co < Cout) {
      float v = acc[r];
      if (bias) v += bias[co];
      size_t oidx = (((size_t)b * Cout + co) * H + y) * W + xo;
      if (resid) v += (float)resid[oidx];
      if (out16) out16[oidx] = (_Float16)v;
      if (out32) out32[oidx] = v;
    }
  }
}

// ---------------- vector quantization ----------------

// z (4,256,64,64) NCHW -> rows [16384][256]
__global__ void k_z_to_rows(const _Float16* __restrict__ z, _Float16* __restrict__ rows,
                            size_t total) {
  for (size_t i = blockIdx.x * (size_t)256 + threadIdx.x; i < total;
       i += (size_t)gridDim.x * 256) {
    int c = (int)(i & 255);
    int row = (int)(i >> 8);
    int b = row >> 12;
    int rem = row & 4095;
    rows[i] = z[((((size_t)b << 8) + c) << 12) + rem];
  }
}

__global__ void k_cb_norm(const float* __restrict__ cb, float* __restrict__ norms) {
  __shared__ float sh[256];
  float v = cb[((size_t)blockIdx.x << 8) + threadIdx.x];
  sh[threadIdx.x] = v * v;
  __syncthreads();
  for (int st = 128; st > 0; st >>= 1) {
    if (threadIdx.x < st) sh[threadIdx.x] += sh[threadIdx.x + st];
    __syncthreads();
  }
  if (threadIdx.x == 0) norms[blockIdx.x] = sh[0];
}

// 16 rows per block; 8 waves sweep all 1024 codes with WMMA dot products,
// tracking running (min, argmin) per D-fragment slot; LDS tree reduce at end.
// The 16x256 f16 x-tile is staged with GLOBAL_LOAD_ASYNC_TO_LDS_B128
// (ASYNCcnt path; INST_OFFSET applies to both the global and LDS address).
__launch_bounds__(256)
__global__ void k_vq_argmin(const _Float16* __restrict__ rows, const _Float16* __restrict__ cb16,
                            const float* __restrict__ enorm, int* __restrict__ idx_out) {
  const int r0 = blockIdx.x << 4;
  __shared__ __align__(16) _Float16 xa[16][256];

  {
    unsigned ldsoff = (unsigned)(uintptr_t)(&xa[0][0]) + (unsigned)threadIdx.x * 16u;
    unsigned long long ga =
        (unsigned long long)(uintptr_t)(rows + ((size_t)r0 << 8)) +
        (unsigned long long)threadIdx.x * 16ull;
    asm volatile("global_load_async_to_lds_b128 %0, %1, off"
                 :: "v"(ldsoff), "v"(ga) : "memory");
    asm volatile("global_load_async_to_lds_b128 %0, %1, off offset:4096"
                 :: "v"(ldsoff), "v"(ga) : "memory");
    asm volatile("s_wait_asynccnt 0x0" ::: "memory");
  }
  __syncthreads();

  const int wave = threadIdx.x >> 5, lane = threadIdx.x & 31;
  float bestv[8];
  int besti[8];
#pragma unroll
  for (int r = 0; r < 8; ++r) { bestv[r] = 3.4e38f; besti[r] = 0; }

  for (int it = 0; it < 8; ++it) {
    int code0 = it * 128 + wave * 16;
    v8f acc = {0.f, 0.f, 0.f, 0.f, 0.f, 0.f, 0.f, 0.f};
    for (int k0 = 0; k0 < 256; k0 += 32) {
      int arow = lane & 15;
      int akb = (lane < 16) ? 0 : 16;
      union { v16h v; v8h h[2]; } au;
      au.h[0] = *(const v8h*)&xa[arow][k0 + akb];
      au.h[1] = *(const v8h*)&xa[arow][k0 + akb + 8];
      int code = code0 + (lane & 15);
      int kc = k0 + ((lane < 16) ? 0 : 16);
      v16h bv = *(const v16h*)(cb16 + ((size_t)code << 8) + kc);
      acc = __builtin_amdgcn_wmma_f32_16x16x32_f16(false, au.v, false, bv,
                                                   (short)0, acc, false, false);
    }
    int code = code0 + (lane & 15);
    float en = enorm[code];
#pragma unroll
    for (int r = 0; r < 8; ++r) {
      float d = en - 2.f * acc[r];
      if (d < bestv[r]) { bestv[r] = d; besti[r] = code; }
    }
  }

  __shared__ float cv[8][8][32];
  __shared__ int   ci[8][8][32];
#pragma unroll
  for (int r = 0; r < 8; ++r) { cv[wave][r][lane] = bestv[r]; ci[wave][r][lane] = besti[r]; }
  __syncthreads();

  int row = threadIdx.x >> 4, j = threadIdx.x & 15;
  int vr = row & 7;
  int ls = (row < 8) ? j : (16 + j);
  float bv = 3.4e38f; int bi = 0;
  for (int w = 0; w < 8; ++w) {
    float v = cv[w][vr][ls];
    if (v < bv) { bv = v; bi = ci[w][vr][ls]; }
  }
  __shared__ float rv[16][16];
  __shared__ int   ri[16][16];
  rv[row][j] = bv; ri[row][j] = bi;
  __syncthreads();
  if (j == 0) {
    float m = rv[row][0]; int mi = ri[row][0];
    for (int t = 1; t < 16; ++t)
      if (rv[row][t] < m) { m = rv[row][t]; mi = ri[row][t]; }
    idx_out[r0 + row] = mi;
  }
}

// zq = codebook[idx], back to NCHW f16; also emit indices.
__global__ void k_vq_gather(const int* __restrict__ idx, const float* __restrict__ cb,
                            _Float16* __restrict__ zq, int* __restrict__ ind_out, size_t total) {
  for (size_t i = blockIdx.x * (size_t)256 + threadIdx.x; i < total;
       i += (size_t)gridDim.x * 256) {
    int c = (int)(i & 255);
    int row = (int)(i >> 8);
    float v = cb[((size_t)idx[row] << 8) + c];
    int b = row >> 12;
    int rem = row & 4095;
    zq[((((size_t)b << 8) + c) << 12) + rem] = (_Float16)v;
    if (c == 0) ind_out[row] = idx[row];
  }
}

// ---------------------------------------------------------------------------
// Host orchestration
// ---------------------------------------------------------------------------
static inline int cdiv_i(size_t a) { return (int)((a + 255) >> 8); }

extern "C" void kernel_launch(void* const* d_in, const int* in_sizes, int n_in,
                              void* d_out, int out_size, void* d_ws, size_t ws_size,
                              hipStream_t stream) {
  (void)in_sizes; (void)n_in; (void)out_size; (void)ws_size;
  auto P = [&](int i) { return (const float*)d_in[i]; };

  char* ws = (char*)d_ws;
  size_t off = 0;
  auto a16 = [&](size_t e) { _Float16* p = (_Float16*)(ws + off); off += ((e * 2 + 255) & ~(size_t)255); return p; };
  auto a32 = [&](size_t e) { float* p = (float*)(ws + off); off += ((e * 4 + 255) & ~(size_t)255); return p; };

  const size_t ACT = 67108864ull;   // max activation: 4 x 256 x 256 x 256
  _Float16* X  = a16(ACT);
  _Float16* S1 = a16(ACT);
  _Float16* S2 = a16(ACT);
  _Float16* Wt = a16((size_t)512 * 512 * 9);
  _Float16* rows = a16((size_t)16384 * 256);
  _Float16* cb16 = a16((size_t)1024 * 256);
  float* stats = a32(4 * 32 * 2);
  float* enorm = a32(1024);
  int* vidx = (int*)a32(16384);

  float* recon = (float*)d_out;
  int* ind_out = (int*)(recon + (size_t)4 * 3 * 256 * 256);

  auto convf = [&](int wi, int bi, const _Float16* in, _Float16* o16, float* o32,
                   const _Float16* res, int Cin, int Cout, int Hh, int Ww, int Ksz) {
    int n = Cout * Cin * Ksz * Ksz;
    k_f32_to_f16<<<(n + 255) / 256, 256, 0, stream>>>(P(wi), Wt, n);
    dim3 g(Hh * (Ww >> 5), (Cout + 63) >> 6, 4);
    k_conv_wmma<<<g, 256, 0, stream>>>(in, Wt, bi >= 0 ? P(bi) : nullptr, res,
                                       o16, o32, Cin, Hh, Ww, Cout, Ksz);
  };
  auto gnsilu = [&](int gi, int bi, const _Float16* in, _Float16* o, int C, int Hh, int Ww) {
    k_gn_stats<<<128, 256, 0, stream>>>(in, stats, C, Hh * Ww);
    size_t total = (size_t)4 * C * Hh * Ww;
    k_gn_silu<<<cdiv_i(total), 256, 0, stream>>>(in, o, P(gi), P(bi), stats, C, Hh * Ww, total);
  };
  auto rb = [&](int c1, int c2, int nin, int n1g, int n1b, int n2g, int n2b,
                int Cin, int Cout, int Hh, int Ww) {
    gnsilu(n1g, n1b, X, S1, Cin, Hh, Ww);
    convf(c1, -1, S1, S2, nullptr, nullptr, Cin, Cout, Hh, Ww, 3);
    gnsilu(n2g, n2b, S2, S1, Cout, Hh, Ww);
    if (nin >= 0) {
      convf(c2, -1, S1, S2, nullptr, nullptr, Cout, Cout, Hh, Ww, 3);
      convf(nin, -1, S2, S1, nullptr, S2, Cout, Cout, Hh, Ww, 1);  // h + nin(h)
      _Float16* t = X; X = S1; S1 = t;
    } else {
      convf(c2, -1, S1, S2, nullptr, X, Cout, Cout, Hh, Ww, 3);     // h + x
      _Float16* t = X; X = S2; S2 = t;
    }
  };
  auto pool = [&](int C, int Hh, int Ww) {
    size_t total = (size_t)4 * C * (Hh >> 1) * (Ww >> 1);
    k_avgpool<<<cdiv_i(total), 256, 0, stream>>>(X, S1, Hh, Ww, total);
    _Float16* t = X; X = S1; S1 = t;
  };
  auto up2 = [&](int C, int Hh, int Ww) {
    size_t total = (size_t)4 * C * Hh * Ww * 4;
    k_up2<<<cdiv_i(total), 256, 0, stream>>>(X, S1, Hh, Ww, total);
    _Float16* t = X; X = S1; S1 = t;
  };

  // ---------------- encoder ----------------
  k_f32_to_f16<<<(786432 + 255) / 256, 256, 0, stream>>>(P(0), X, 786432);
  convf(62, -1, X, S1, nullptr, nullptr, 3, 128, 256, 256, 3); { _Float16* t = X; X = S1; S1 = t; }
  rb(65, 66, -1, 67, 68, 69, 70, 128, 128, 256, 256);
  rb(71, 72, -1, 73, 74, 75, 76, 128, 128, 256, 256);
  pool(128, 256, 256);
  rb(77, 78, 79, 80, 81, 82, 83, 128, 256, 128, 128);
  rb(84, 85, -1, 86, 87, 88, 89, 256, 256, 128, 128);
  pool(256, 128, 128);
  rb(90, 91, 92, 93, 94, 95, 96, 256, 512, 64, 64);
  rb(97, 98, -1, 99, 100, 101, 102, 512, 512, 64, 64);
  rb(103, 104, -1, 105, 106, 107, 108, 512, 512, 64, 64);   // mid[0]
  rb(109, 110, -1, 111, 112, 113, 114, 512, 512, 64, 64);   // mid[1]
  gnsilu(115, 116, X, S1, 512, 64, 64);
  convf(64, 63, S1, S2, nullptr, nullptr, 512, 256, 64, 64, 1); { _Float16* t = X; X = S2; S2 = t; }

  // ---------------- quantize ----------------
  { size_t total = (size_t)16384 * 256;
    k_z_to_rows<<<cdiv_i(total), 256, 0, stream>>>(X, rows, total); }
  k_f32_to_f16<<<(1024 * 256 + 255) / 256, 256, 0, stream>>>(P(1), cb16, 1024 * 256);
  k_cb_norm<<<1024, 256, 0, stream>>>(P(1), enorm);
  k_vq_argmin<<<1024, 256, 0, stream>>>(rows, cb16, enorm, vidx);
  { size_t total = (size_t)16384 * 256;
    k_vq_gather<<<cdiv_i(total), 256, 0, stream>>>(vidx, P(1), S1, ind_out, total); }
  { _Float16* t = X; X = S1; S1 = t; }

  // ---------------- decoder ----------------
  convf(3, 2, X, S1, nullptr, nullptr, 256, 512, 64, 64, 3); { _Float16* t = X; X = S1; S1 = t; }
  rb(6, 7, -1, 8, 9, 10, 11, 512, 512, 64, 64);              // mid[0]
  rb(12, 13, -1, 14, 15, 16, 17, 512, 512, 64, 64);          // mid[1]
  // up[2]
  rb(48, 49, -1, 50, 51, 52, 53, 512, 512, 64, 64);
  rb(54, 55, -1, 56, 57, 58, 59, 512, 512, 64, 64);
  up2(512, 64, 64);
  convf(61, 60, X, S1, nullptr, nullptr, 512, 512, 128, 128, 3); { _Float16* t = X; X = S1; S1 = t; }
  // up[1]
  rb(33, 34, 35, 36, 37, 38, 39, 512, 256, 128, 128);
  rb(40, 41, -1, 42, 43, 44, 45, 256, 256, 128, 128);
  up2(256, 128, 128);
  convf(47, 46, X, S1, nullptr, nullptr, 256, 256, 256, 256, 3); { _Float16* t = X; X = S1; S1 = t; }
  // up[0]
  rb(20, 21, 22, 23, 24, 25, 26, 256, 128, 256, 256);
  rb(27, 28, -1, 29, 30, 31, 32, 128, 128, 256, 256);
  gnsilu(18, 19, X, S1, 128, 256, 256);
  convf(5, 4, S1, nullptr, recon, nullptr, 128, 3, 256, 256, 3);
}
